// ResidualGNNV3_88914412961893
// MI455X (gfx1250) — compile-verified, matrix operand order checked by reading, not run on hardware
//
#include <hip/hip_runtime.h>
#include <math.h>
#include <stdint.h>

#define DIM   128
#define H2DIM 64
#define CDIM  3
#define EPSLN 1e-5f

typedef __attribute__((ext_vector_type(16))) __bf16 v16bf;
typedef __attribute__((ext_vector_type(8)))  float  v8f;

// K index inside a 16x32 bf16 A/B fragment, per CDNA5 ISA 7.12.2.
// element e (0..15) of the per-lane v16bf; hi = (lane >= 16)
__device__ __forceinline__ int kmap(int e, int hi) {
    int v = e >> 1, h = e & 1;
    return ((v >> 2) << 4) | ((v & 3) << 1) | h | (hi << 3);
}

// ---------------- utility fills ----------------
__global__ void k_fill(float* __restrict__ p, int n, float v) {
    int i = blockIdx.x * blockDim.x + threadIdx.x;
    if (i < n) p[i] = v;
}

// ---------------- degree / norm ----------------
__global__ void k_degree(const long long* __restrict__ dst, int E, float* __restrict__ deg) {
    int i = blockIdx.x * blockDim.x + threadIdx.x;
    if (i < E) atomicAdd(&deg[(int)dst[i]], 1.0f);
}

__global__ void k_rsqrt(float* __restrict__ d, int n) {
    int i = blockIdx.x * blockDim.x + threadIdx.x;
    if (i < n) d[i] = rsqrtf(d[i]);   // deg >= 1 always (self loop)
}

// ---------------- weight pre-pack: fp32 W[128x128] -> fragment-layout bf16 hi/lo ----
// layout index: (((kt*8 + nt)*32 + lane)*16 + e), kt = k0/32, nt = n0/16
// so each lane's 16 fragment elements are 32 contiguous bytes (2x b128, coalesced).
__global__ void k_pack_w(const float* __restrict__ W,
                         __bf16* __restrict__ Whi, __bf16* __restrict__ Wlo) {
    int idx = blockIdx.x * blockDim.x + threadIdx.x;   // 4*8*32*16 = 16384
    if (idx >= 4 * 8 * 32 * 16) return;
    int e  = idx & 15;
    int L  = (idx >> 4) & 31;
    int nt = (idx >> 9) & 7;
    int kt = idx >> 12;
    int k  = kt * 32 + kmap(e, L >> 4);
    int n  = nt * 16 + (L & 15);
    float f = W[(size_t)k * DIM + n];
    __bf16 h = (__bf16)f;
    Whi[idx] = h;
    Wlo[idx] = (__bf16)(f - (float)h);
}

// ---------------- GEMM: Out[n x 128] = A[n x 128] @ W[128 x 128] ----------------
// bf16 hi/lo split (3-term) for ~fp32 accuracy, fp32 accumulate in WMMA.
// Block = 256 threads = 8 waves; block owns 16 rows; wave w owns cols [16w,16w+16).
// A tile is DMA'd to LDS with GLOBAL_LOAD_ASYNC_TO_LDS (ASYNCcnt path, no VGPR
// round-trip); fragments are built ONCE per block; W fragments come pre-packed.
__global__ __launch_bounds__(256) void k_gemm_ws(
    const float* __restrict__ A,
    const __bf16* __restrict__ Whi, const __bf16* __restrict__ Wlo,
    float* __restrict__ Out, int nrows) {
    __shared__ __align__(16) float As[16][DIM + 4];   // +4: rows stay 16B-aligned,
                                                      // bank stride 4 -> conflict-free
    __shared__ __align__(32) __bf16 AfH[4][32][16];   // fragment-layout A (hi)
    __shared__ __align__(32) __bf16 AfL[4][32][16];   // fragment-layout A (lo)
    const int t = threadIdx.x;
    const int rowBase = blockIdx.x * 16;

    // async-copy the 16x128 fp32 tile as 512 x 16B chunks straight into LDS
    #pragma unroll
    for (int q = 0; q < 2; ++q) {
        int i  = t + q * 256;                  // chunk id
        int r  = i >> 5;                       // row 0..15
        int cc = (i & 31) << 2;                // col (floats), 16B chunks
        int gr = rowBase + r;
        if (gr >= nrows) gr = nrows - 1;       // clamp: OOB C-rows are never stored
        unsigned lds = (unsigned)(uintptr_t)&As[r][cc];
        unsigned long long ga =
            (unsigned long long)(uintptr_t)(A + (size_t)gr * DIM + cc);
        asm volatile("global_load_async_to_lds_b128 %0, %1, off"
                     :: "v"(lds), "v"(ga) : "memory");
    }
    asm volatile("s_wait_asynccnt 0x0" ::: "memory");
    __syncthreads();

    #pragma unroll
    for (int q = 0; q < 8; ++q) {                     // build A fragments once
        int idx = t * 8 + q;                          // 4*32*16 = 2048 entries
        int e  = idx & 15;
        int L  = (idx >> 4) & 31;
        int kt = idx >> 9;
        int m  = L & 15;
        int k  = kt * 32 + kmap(e, L >> 4);
        float f = As[m][k];
        __bf16 h = (__bf16)f;
        AfH[kt][L][e] = h;
        AfL[kt][L][e] = (__bf16)(f - (float)h);
    }
    __syncthreads();

    const int lane = t & 31;
    const int wv   = t >> 5;                          // wave = output col tile
    v8f c = {};
    #pragma unroll
    for (int kt = 0; kt < 4; ++kt) {
        v16bf ahi = *(const v16bf*)&AfH[kt][lane][0]; // 2x ds_load_b128
        v16bf alo = *(const v16bf*)&AfL[kt][lane][0];
        size_t boff = (((size_t)kt * 8 + wv) * 32 + lane) * 16;
        v16bf bhi = *(const v16bf*)(Whi + boff);      // 2x global_load_b128
        v16bf blo = *(const v16bf*)(Wlo + boff);
        c = __builtin_amdgcn_wmma_f32_16x16x32_bf16(false, ahi, false, bhi, (short)0, c, false, false);
        c = __builtin_amdgcn_wmma_f32_16x16x32_bf16(false, ahi, false, blo, (short)0, c, false, false);
        c = __builtin_amdgcn_wmma_f32_16x16x32_bf16(false, alo, false, bhi, (short)0, c, false, false);
    }

    const int l16 = lane & 15, hi = lane >> 4, n0 = wv * 16;
    #pragma unroll
    for (int r = 0; r < 8; ++r) {                     // D: M = r + 8*hi, N = lane%16
        int row = rowBase + r + hi * 8;
        if (row < nrows) Out[(size_t)row * DIM + n0 + l16] = c[r];
    }
}

// ---------------- edge scatter: agg[dst] += hh[src] * dinv[src]*dinv[dst] ----------------
// one wave32 per edge (edges E.. are the N self-loops), float4 per lane
__global__ __launch_bounds__(256) void k_scatter(
    const float* __restrict__ hh, float* __restrict__ agg,
    const long long* __restrict__ src, const long long* __restrict__ dst,
    const float* __restrict__ dinv, int E, int n) {
    int gw = (blockIdx.x * blockDim.x + threadIdx.x) >> 5;
    int lane = threadIdx.x & 31;
    if (gw >= E + n) return;
    int s, d;
    if (gw < E) { s = (int)src[gw]; d = (int)dst[gw]; }
    else        { s = d = gw - E; }
    float nrm = dinv[s] * dinv[d];
    const float4 v = *(const float4*)(hh + (size_t)s * DIM + lane * 4);
    float* ap = agg + (size_t)d * DIM + lane * 4;
    atomicAdd(ap + 0, v.x * nrm);
    atomicAdd(ap + 1, v.y * nrm);
    atomicAdd(ap + 2, v.z * nrm);
    atomicAdd(ap + 3, v.w * nrm);
}

// ---------------- bias + layernorm + relu, one wave per node ----------------
__global__ __launch_bounds__(256) void k_bias_ln_relu(
    const float* __restrict__ agg, const float* __restrict__ bias,
    const float* __restrict__ lw, const float* __restrict__ lb,
    float* __restrict__ out, int n) {
    int gw = (blockIdx.x * blockDim.x + threadIdx.x) >> 5;
    int lane = threadIdx.x & 31;
    if (gw >= n) return;
    const float4 a = *(const float4*)(agg + (size_t)gw * DIM + lane * 4);
    const float4 b = *(const float4*)(bias + lane * 4);
    float4 v = {a.x + b.x, a.y + b.y, a.z + b.z, a.w + b.w};
    float s  = v.x + v.y + v.z + v.w;
    float sq = v.x*v.x + v.y*v.y + v.z*v.z + v.w*v.w;
    for (int off = 16; off > 0; off >>= 1) {    // wave32 tree reduce
        s  += __shfl_xor(s,  off, 32);
        sq += __shfl_xor(sq, off, 32);
    }
    float mu   = s * (1.0f / DIM);
    float var  = sq * (1.0f / DIM) - mu * mu;
    float rstd = rsqrtf(var + EPSLN);
    const float4 w  = *(const float4*)(lw + lane * 4);
    const float4 lv = *(const float4*)(lb + lane * 4);
    float4 o;
    o.x = fmaxf((v.x - mu) * rstd * w.x + lv.x, 0.0f);
    o.y = fmaxf((v.y - mu) * rstd * w.y + lv.y, 0.0f);
    o.z = fmaxf((v.z - mu) * rstd * w.z + lv.z, 0.0f);
    o.w = fmaxf((v.w - mu) * rstd * w.w + lv.w, 0.0f);
    *(float4*)(out + (size_t)gw * DIM + lane * 4) = o;
}

// ---------------- column sum for mean pool ----------------
#define MEAN_ROWS 256
__global__ __launch_bounds__(256) void k_colsum(const float* __restrict__ h,
                                                float* __restrict__ g, int n) {
    __shared__ float s[256];
    int t = threadIdx.x;
    int f = t & 127, half = t >> 7;
    int base = blockIdx.x * MEAN_ROWS;
    float acc = 0.0f;
    for (int r = half; r < MEAN_ROWS; r += 2) {
        int node = base + r;
        if (node < n) {
            __builtin_prefetch(&h[(size_t)(node + 16) * DIM + f], 0, 0); // global_prefetch_b8
            acc += h[(size_t)node * DIM + f];
        }
    }
    s[t] = acc;
    __syncthreads();
    if (t < 128) atomicAdd(&g[f], s[t] + s[t + 128]);
}

// ---------------- gproj = (g/N) @ Wg + bg  (tiny, one block of 128) ----------------
__global__ void k_gproj(const float* __restrict__ g, const float* __restrict__ Wg,
                        const float* __restrict__ bg, float* __restrict__ gp, float invn) {
    int j = threadIdx.x;
    float acc = bg[j];
    for (int k = 0; k < DIM; ++k) acc += g[k] * invn * Wg[(size_t)k * DIM + j];
    gp[j] = acc;
}

// ---------------- MLP head: out = tanh(relu((h+gp)@Wf1 + bf1) @ Wf2 + bf2) ----------------
// one wave per node, staged through LDS
__global__ __launch_bounds__(256) void k_head(
    const float* __restrict__ h, const float* __restrict__ gp,
    const float* __restrict__ Wf1, const float* __restrict__ bf1,
    const float* __restrict__ Wf2, const float* __restrict__ bf2,
    float* __restrict__ out, int n) {
    __shared__ float hv[8][DIM];
    __shared__ float t1[8][H2DIM];
    int t = threadIdx.x, wave = t >> 5, lane = t & 31;
    int node = blockIdx.x * 8 + wave;
    bool act = node < n;
    if (act) {
        const float4 a = *(const float4*)(h + (size_t)node * DIM + lane * 4);
        const float4 b = *(const float4*)(gp + lane * 4);
        float4 v = {a.x + b.x, a.y + b.y, a.z + b.z, a.w + b.w};
        *(float4*)(&hv[wave][lane * 4]) = v;
    }
    __syncthreads();
    if (act) {
        #pragma unroll
        for (int jj = 0; jj < 2; ++jj) {
            int j = lane + jj * 32;
            float acc = bf1[j];
            for (int k = 0; k < DIM; ++k) acc += hv[wave][k] * Wf1[(size_t)k * H2DIM + j];
            t1[wave][j] = fmaxf(acc, 0.0f);
        }
    }
    __syncthreads();
    if (act && lane < CDIM) {
        float acc = bf2[lane];
        for (int k = 0; k < H2DIM; ++k) acc += t1[wave][k] * Wf2[(size_t)k * CDIM + lane];
        out[(size_t)node * CDIM + lane] = tanhf(acc);
    }
}

extern "C" void kernel_launch(void* const* d_in, const int* in_sizes, int n_in,
                              void* d_out, int out_size, void* d_ws, size_t ws_size,
                              hipStream_t stream) {
    const float*     x    = (const float*)d_in[0];
    const long long* ei   = (const long long*)d_in[1];
    const float*     W1   = (const float*)d_in[2];
    const float*     b1   = (const float*)d_in[3];
    const float*     ln1w = (const float*)d_in[4];
    const float*     ln1b = (const float*)d_in[5];
    const float*     W2   = (const float*)d_in[6];
    const float*     b2   = (const float*)d_in[7];
    const float*     ln2w = (const float*)d_in[8];
    const float*     ln2b = (const float*)d_in[9];
    const float*     Wg   = (const float*)d_in[10];
    const float*     bg   = (const float*)d_in[11];
    const float*     Wf1  = (const float*)d_in[12];
    const float*     bf1  = (const float*)d_in[13];
    const float*     Wf2  = (const float*)d_in[14];
    const float*     bf2  = (const float*)d_in[15];

    const int n = in_sizes[0] / DIM;       // 50000
    const int E = in_sizes[1] / 2;         // 600000
    const long long* src = ei;
    const long long* dst = ei + E;

    // workspace: dinv[n] | bufA[n*DIM] | bufB[n*DIM] | g | gp | packed weights (~51.6 MB)
    float* ws   = (float*)d_ws;
    float* dinv = ws;
    float* bufA = ws + ((n + 63) & ~63);
    float* bufB = bufA + (size_t)n * DIM;
    float* g    = bufB + (size_t)n * DIM;
    float* gp   = g + DIM;
    const int PACKN = 4 * 8 * 32 * 16;     // 16384 bf16 per plane
    __bf16* W1h = (__bf16*)(gp + DIM);
    __bf16* W1l = W1h + PACKN;
    __bf16* W2h = W1l + PACKN;
    __bf16* W2l = W2h + PACKN;

    const int ND = n * DIM;
    dim3 blk(256);
    auto cdiv = [](int a, int b) { return (a + b - 1) / b; };

    // pre-pack weights into WMMA fragment layout (bf16 hi/lo planes)
    k_pack_w<<<cdiv(PACKN, 256), blk, 0, stream>>>(W1, W1h, W1l);
    k_pack_w<<<cdiv(PACKN, 256), blk, 0, stream>>>(W2, W2h, W2l);

    // degrees -> dinv
    k_fill<<<cdiv(n, 256), blk, 0, stream>>>(dinv, n, 1.0f);       // self loops
    k_degree<<<cdiv(E, 256), blk, 0, stream>>>(dst, E, dinv);
    k_rsqrt<<<cdiv(n, 256), blk, 0, stream>>>(dinv, n);

    // ---- layer 1: bufA = x@W1 ; bufB = scatter(bufA) ; bufA = relu(ln(bufB+b1)) ----
    k_gemm_ws<<<cdiv(n, 16), blk, 0, stream>>>(x, W1h, W1l, bufA, n);
    k_fill<<<cdiv(ND, 256), blk, 0, stream>>>(bufB, ND, 0.0f);
    k_scatter<<<cdiv((E + n) * 32, 256), blk, 0, stream>>>(bufA, bufB, src, dst, dinv, E, n);
    k_bias_ln_relu<<<cdiv(n * 32, 256), blk, 0, stream>>>(bufB, b1, ln1w, ln1b, bufA, n);

    // ---- layer 2: bufB = bufA@W2 ; bufA = scatter(bufB) ; bufB = relu(ln(bufA+b2)) ----
    k_gemm_ws<<<cdiv(n, 16), blk, 0, stream>>>(bufA, W2h, W2l, bufB, n);
    k_fill<<<cdiv(ND, 256), blk, 0, stream>>>(bufA, ND, 0.0f);
    k_scatter<<<cdiv((E + n) * 32, 256), blk, 0, stream>>>(bufB, bufA, src, dst, dinv, E, n);
    k_bias_ln_relu<<<cdiv(n * 32, 256), blk, 0, stream>>>(bufA, b2, ln2w, ln2b, bufB, n);

    // ---- global mean pool + broadcast projection ----
    k_fill<<<1, 128, 0, stream>>>(g, DIM, 0.0f);
    k_colsum<<<cdiv(n, MEAN_ROWS), blk, 0, stream>>>(bufB, g, n);
    k_gproj<<<1, 128, 0, stream>>>(g, Wg, bg, gp, 1.0f / (float)n);

    // ---- MLP head ----
    k_head<<<cdiv(n * 32, 256), blk, 0, stream>>>(bufB, gp, Wf1, bf1, Wf2, bf2,
                                                  (float*)d_out, n);
}
